// CustomGPT2MultiHeadAttention_22600117911808
// MI455X (gfx1250) — matrix operand order repaired and checked by reading, not compiled, not run on hardware
//
#include <hip/hip_runtime.h>
#include <stdint.h>

typedef __bf16 bf16_t;
typedef bf16_t v16bf __attribute__((ext_vector_type(16)));
typedef bf16_t v8bf  __attribute__((ext_vector_type(8)));
typedef float  v8f   __attribute__((ext_vector_type(8)));
typedef unsigned int u32;
typedef unsigned long long u64;
typedef u32 v4u __attribute__((ext_vector_type(4)));
typedef int  v8i __attribute__((ext_vector_type(8)));
typedef int  v4i __attribute__((ext_vector_type(4)));

namespace {
constexpr int kB = 8, kS = 1024, kD = 1024, kH = 16, kHD = 64;
constexpr int kKLen = kS + 1;  // attention_mask row length (text + 1 image token)
}

// Build a 16-element bf16 WMMA operand from two 16-byte contiguous chunks.
static __device__ __forceinline__ v16bf ld2x8(const bf16_t* p0, const bf16_t* p1) {
  v8bf lo = *(const v8bf*)p0;
  v8bf hi = *(const v8bf*)p1;
  v16bf r;
#pragma unroll
  for (int i = 0; i < 8; ++i) { r[i] = lo[i]; r[i + 8] = hi[i]; }
  return r;
}

static __device__ __forceinline__ v8f wmma_bf(v16bf a, v16bf b, v8f c) {
  // D(f32 16x16) = A(16x32 bf16) * B(32x16 bf16) + C
  return __builtin_amdgcn_wmma_f32_16x16x32_bf16(false, a, false, b, (short)0, c,
                                                 false, false);
}

// ---- Tensor Data Mover: 2D tile (bf16) global -> LDS (D# per ISA 08 §8.3/8.4) ----
static __device__ __forceinline__ void tdm_load_2d(u32 lds_byte_addr, const void* gptr,
                                                   u32 dim0, u32 dim1, u32 stride0,
                                                   u32 tile0, u32 tile1) {
  u64 ga = (u64)(uintptr_t)gptr;
  v4u g0;
  g0.x = 1u;                              // count=1 (valid user descriptor)
  g0.y = lds_byte_addr;                   // lds_addr [63:32]
  g0.z = (u32)ga;                         // global_addr [95:64]
  g0.w = (u32)(ga >> 32) | (2u << 30);    // global_addr[56:32] | type=2 ("image")
  v8i g1;
  g1[0] = 0x00010000;                                   // data_size=1 -> 2 bytes
  g1[1] = (int)((dim0 & 0xffffu) << 16);                // tensor_dim0[15:0]
  g1[2] = (int)((dim0 >> 16) | ((dim1 & 0xffffu) << 16));  // dim0 hi | dim1 lo
  g1[3] = (int)((dim1 >> 16) | (tile0 << 16));          // dim1 hi | tile_dim0
  g1[4] = (int)(tile1 & 0xffffu);                       // tile_dim1 (tile_dim2=0)
  g1[5] = (int)stride0;                                 // tensor_dim0_stride lo
  g1[6] = 0;                                            // stride0 hi | dim1_stride lo
  g1[7] = 0;
  v4i z = {0, 0, 0, 0};
#if defined(__clang_major__) && (__clang_major__ >= 23)
  v8i z8 = {0, 0, 0, 0, 0, 0, 0, 0};
  __builtin_amdgcn_tensor_load_to_lds(g0, g1, z, z, z8, 0);
#else
  __builtin_amdgcn_tensor_load_to_lds(g0, g1, z, z, 0);
#endif
}

// ---------------- conversion kernels ----------------
__global__ __launch_bounds__(256) void cvt_f32_bf16(const float* __restrict__ in,
                                                    bf16_t* __restrict__ out, int n) {
  for (int i = blockIdx.x * blockDim.x + threadIdx.x; i < n;
       i += gridDim.x * blockDim.x)
    out[i] = (bf16_t)in[i];
}

// in[R][C] (f32) -> out[C][R] (bf16): lets GEMM B-tiles read contiguous K per lane.
__global__ __launch_bounds__(256) void cvt_transpose_bf16(const float* __restrict__ in,
                                                          bf16_t* __restrict__ out,
                                                          int R, int C) {
  int n = R * C;
  for (int i = blockIdx.x * blockDim.x + threadIdx.x; i < n;
       i += gridDim.x * blockDim.x) {
    int cc = i / R, rr = i - cc * R;
    out[i] = (bf16_t)in[rr * C + cc];
  }
}

// ---------------- QKV projection GEMM ----------------
// X[8192,1024] (bf16) @ Wat^T (Wat stored [3072][1024]) + bias -> scatter to
// Q[B,H,S,64] (pre-scaled by 1/8), K[B,H,S,64], Vt[B,H,64,S].
__global__ __launch_bounds__(128) void gemm_qkv(const bf16_t* __restrict__ Xb,
                                                const bf16_t* __restrict__ Wat,
                                                const float* __restrict__ bias,
                                                bf16_t* __restrict__ Qb,
                                                bf16_t* __restrict__ Kb,
                                                bf16_t* __restrict__ Vtb) {
  const int lane = threadIdx.x & 31;
  const int wave = threadIdx.x >> 5;
  const int r = lane & 15;     // A: row / B: column / D: column
  const int hh = lane >> 4;    // lane half selects K sub-chunk / D row group
  const int m0 = blockIdx.x * 16;
  const int n0 = blockIdx.y * 256 + wave * 64;  // 64-aligned: one head per wave

  v8f acc[4] = {};
  const bf16_t* arow = Xb + (size_t)(m0 + r) * kD;
#pragma unroll 1
  for (int k0 = 0; k0 < kD; k0 += 32) {
    v16bf a = ld2x8(arow + k0 + 8 * hh, arow + k0 + 16 + 8 * hh);
#pragma unroll
    for (int t = 0; t < 4; ++t) {
      const bf16_t* bp = Wat + (size_t)(n0 + 16 * t + r) * kD + k0 + 8 * hh;
      acc[t] = wmma_bf(a, ld2x8(bp, bp + 16), acc[t]);
    }
  }

  // n0 is wave-uniform: force scalar so Q/K/V dispatch is s_cbranch, not exec masks.
  const int n0s = __builtin_amdgcn_readfirstlane(n0);
  const int region = n0s >> 10;        // 0=Q 1=K 2=V
  const int head = (n0s & 1023) >> 6;
  const int bb = m0 >> 10;
  const int s0 = m0 & 1023;
#pragma unroll
  for (int t = 0; t < 4; ++t) {
    const int hd = 16 * t + r;
    const float bv = bias[n0 + 16 * t + r];
#pragma unroll
    for (int j = 0; j < 8; ++j) {
      float v = acc[t][j] + bv;
      int s = s0 + j + 8 * hh;
      size_t bhs = (size_t)(bb * kH + head) * kS + s;
      if (region == 0)
        Qb[bhs * kHD + hd] = (bf16_t)(v * 0.125f);  // fold 1/sqrt(64)
      else if (region == 1)
        Kb[bhs * kHD + hd] = (bf16_t)v;
      else
        Vtb[((size_t)(bb * kH + head) * kHD + hd) * kS + s] = (bf16_t)v;
    }
  }
}

// ---------------- causal flash attention ----------------
// One wave per (b,h,q-tile of 16). Online softmax, f32 accumulation.
// K/V tiles staged global->LDS by the Tensor Data Mover, double-buffered so the
// next tile pair's DMA overlaps this tile's softmax + WMMA (TENSORcnt in-order).
// Per-wave LDS (elements): K bufs [0,2048)+[2048,4096), V bufs [4096,6144)+[6144,8192),
// P tile [8192,8704). 17 KB/wave, 2 waves/block.
__global__ __launch_bounds__(64) void flash_attn(const bf16_t* __restrict__ Qb,
                                                 const bf16_t* __restrict__ Kb,
                                                 const bf16_t* __restrict__ Vtb,
                                                 const float* __restrict__ amask,
                                                 bf16_t* __restrict__ Ab) {
  __shared__ __align__(16) bf16_t smem[2][8704];
  const int lane = threadIdx.x & 31;
  const int wave = threadIdx.x >> 5;
  const int r = lane & 15;
  const int hh = lane >> 4;
  const int gid = blockIdx.x * 2 + wave;
  const int bh = gid >> 6;
  const int q0 = (gid & 63) << 4;
  const int bb = bh >> 4;
  const int hd0 = (bh & 15) * kHD;

  const bf16_t* Qh = Qb + (size_t)bh * kS * kHD;
  const bf16_t* Kh = Kb + (size_t)bh * kS * kHD;
  const bf16_t* Vh = Vtb + (size_t)bh * kHD * kS;
  const float* mrow = amask + (size_t)bb * kKLen;

  bf16_t* smw = &smem[wave][0];
  const u32 lbase = (u32)(uintptr_t)smw;  // LDS byte address (generic low 32 bits)

  const bf16_t* qp = Qh + (size_t)(q0 + r) * kHD;
  v16bf aq0 = ld2x8(qp + 8 * hh, qp + 16 + 8 * hh);        // hd 0..31
  v16bf aq1 = ld2x8(qp + 32 + 8 * hh, qp + 48 + 8 * hh);   // hd 32..63

  v8f o[4] = {};
  float m8[8], l8[8];
#pragma unroll
  for (int j = 0; j < 8; ++j) { m8[j] = -3.0e38f; l8[j] = 0.f; }

  const int kend = q0 + 15;
  const int nsteps = kend / 32 + 1;

  // Prologue: DMA first K (32x64) and V^T (64x32) tiles into buffer 0.
  tdm_load_2d(lbase + 0,    Kh, /*dim0*/64,  /*dim1*/1024, /*stride0*/64,  /*tile0*/64, /*tile1*/32);
  tdm_load_2d(lbase + 8192, Vh, /*dim0*/1024,/*dim1*/64,   /*stride0*/1024,/*tile0*/32, /*tile1*/64);

  int cur = 0;
  for (int it = 0; it < nsteps; ++it) {
    const int k0 = it * 32;
    if (it + 1 < nsteps) {
      const int kn = k0 + 32;
      asm volatile("s_wait_dscnt 0" ::: "memory");  // prior LDS reads done before DMA reuses buffers
      tdm_load_2d(lbase + (u32)((cur ^ 1) * 4096),        Kh + (size_t)kn * kHD,
                  64, 1024, 64, 64, 32);
      tdm_load_2d(lbase + 8192u + (u32)((cur ^ 1) * 4096), Vh + kn,
                  1024, 64, 1024, 32, 64);
      __builtin_amdgcn_s_wait_tensorcnt(2);  // current pair landed; next pair in flight
    } else {
      __builtin_amdgcn_s_wait_tensorcnt(0);
    }
    asm volatile("" ::: "memory");

    // S = Q K^T from LDS tile (rows = key positions, 64 bf16 each)
    const bf16_t* kb = smw + (size_t)cur * 2048;
    const bf16_t* kp0 = kb + r * 64 + 8 * hh;
    const bf16_t* kp1 = kb + (16 + r) * 64 + 8 * hh;
    v8f sc0 = {}, sc1 = {};
    sc0 = wmma_bf(aq0, ld2x8(kp0, kp0 + 16), sc0);
    sc0 = wmma_bf(aq1, ld2x8(kp0 + 32, kp0 + 48), sc0);
    sc1 = wmma_bf(aq0, ld2x8(kp1, kp1 + 16), sc1);
    sc1 = wmma_bf(aq1, ld2x8(kp1 + 32, kp1 + 48), sc1);

    const float mk0 = mrow[k0 + r];
    const float mk1 = mrow[k0 + 16 + r];
#pragma unroll
    for (int j = 0; j < 8; ++j) {
      const int R = q0 + j + 8 * hh;
      float x0 = sc0[j];
      float x1 = sc1[j];
      if (k0 + r > R) x0 = -1.0e9f;        // causal (image token j==1024 never visible)
      if (k0 + 16 + r > R) x1 = -1.0e9f;
      if (mk0 == 0.f) x0 = -1.0e4f;        // attention_mask overrides (ref order)
      if (mk1 == 0.f) x1 = -1.0e4f;
      float mx = fmaxf(x0, x1);
      mx = fmaxf(mx, __shfl_xor(mx, 1, 32));
      mx = fmaxf(mx, __shfl_xor(mx, 2, 32));
      mx = fmaxf(mx, __shfl_xor(mx, 4, 32));
      mx = fmaxf(mx, __shfl_xor(mx, 8, 32));
      const float mnew = fmaxf(m8[j], mx);
      const float al = __expf(m8[j] - mnew);
      const float p0 = __expf(x0 - mnew);
      const float p1 = __expf(x1 - mnew);
      float rs = p0 + p1;
      rs += __shfl_xor(rs, 1, 32);
      rs += __shfl_xor(rs, 2, 32);
      rs += __shfl_xor(rs, 4, 32);
      rs += __shfl_xor(rs, 8, 32);
      l8[j] = l8[j] * al + rs;
      m8[j] = mnew;
      o[0][j] *= al; o[1][j] *= al; o[2][j] *= al; o[3][j] *= al;
      // D-layout -> row-major P tile in LDS
      smw[8192 + (j + 8 * hh) * 32 + r] = (bf16_t)p0;
      smw[8192 + (j + 8 * hh) * 32 + 16 + r] = (bf16_t)p1;
    }
    asm volatile("s_wait_dscnt 0" ::: "memory");
    // Reload P in A-layout (16x32, contraction over 32 key positions)
    const bf16_t* pb = smw + 8192;
    v16bf pa = ld2x8(pb + r * 32 + 8 * hh, pb + r * 32 + 16 + 8 * hh);
    const bf16_t* vb = smw + 4096 + (size_t)cur * 2048;  // 64 rows (hd) x 32 (k)
#pragma unroll
    for (int t = 0; t < 4; ++t) {
      const bf16_t* vp = vb + (16 * t + r) * 32 + 8 * hh;
      o[t] = wmma_bf(pa, ld2x8(vp, vp + 16), o[t]);
    }
    cur ^= 1;
  }

#pragma unroll
  for (int t = 0; t < 4; ++t) {
#pragma unroll
    for (int j = 0; j < 8; ++j) {
      float v = o[t][j] / l8[j];
      int row = q0 + j + 8 * hh;
      Ab[((size_t)bb * kS + row) * kD + hd0 + 16 * t + r] = (bf16_t)v;
    }
  }
}

// ---------------- output projection GEMM ----------------
__global__ __launch_bounds__(128) void gemm_proj(const bf16_t* __restrict__ Ain,
                                                 const bf16_t* __restrict__ Wpt,
                                                 const float* __restrict__ bias,
                                                 float* __restrict__ out) {
  const int lane = threadIdx.x & 31;
  const int wave = threadIdx.x >> 5;
  const int r = lane & 15;
  const int hh = lane >> 4;
  const int m0 = blockIdx.x * 16;
  const int n0 = blockIdx.y * 256 + wave * 64;

  v8f acc[4] = {};
  const bf16_t* arow = Ain + (size_t)(m0 + r) * kD;
#pragma unroll 1
  for (int k0 = 0; k0 < kD; k0 += 32) {
    v16bf a = ld2x8(arow + k0 + 8 * hh, arow + k0 + 16 + 8 * hh);
#pragma unroll
    for (int t = 0; t < 4; ++t) {
      const bf16_t* bp = Wpt + (size_t)(n0 + 16 * t + r) * kD + k0 + 8 * hh;
      acc[t] = wmma_bf(a, ld2x8(bp, bp + 16), acc[t]);
    }
  }
#pragma unroll
  for (int t = 0; t < 4; ++t) {
    const float bv = bias[n0 + 16 * t + r];
#pragma unroll
    for (int j = 0; j < 8; ++j)
      out[(size_t)(m0 + j + 8 * hh) * kD + n0 + 16 * t + r] = acc[t][j] + bv;
  }
}

extern "C" void kernel_launch(void* const* d_in, const int* in_sizes, int n_in,
                              void* d_out, int out_size, void* d_ws, size_t ws_size,
                              hipStream_t stream) {
  (void)in_sizes; (void)n_in; (void)out_size; (void)ws_size;
  const float* hs      = (const float*)d_in[0];
  const float* amask   = (const float*)d_in[1];
  // d_in[2] image_hidden_states, d_in[7] Wuk, d_in[8] Wuv: the causal tril mask
  // makes key position 1024 invisible to every query (j<=i, i<=1023), so the
  // image-KV branch contributes exactly zero to the output — skipped.
  const float* Wc_attn = (const float*)d_in[3];
  const float* bc_attn = (const float*)d_in[4];
  const float* Wc_proj = (const float*)d_in[5];
  const float* bc_proj = (const float*)d_in[6];
  float* out = (float*)d_out;

  char* ws = (char*)d_ws;
  size_t off = 0;
  bf16_t* Xb  = (bf16_t*)(ws + off); off += (size_t)kB * kS * kD * 2;
  bf16_t* Wat = (bf16_t*)(ws + off); off += (size_t)3 * kD * kD * 2;
  bf16_t* Wpt = (bf16_t*)(ws + off); off += (size_t)kD * kD * 2;
  bf16_t* Qb  = (bf16_t*)(ws + off); off += (size_t)kB * kS * kD * 2;
  bf16_t* Kb  = (bf16_t*)(ws + off); off += (size_t)kB * kS * kD * 2;
  bf16_t* Vtb = (bf16_t*)(ws + off); off += (size_t)kB * kS * kD * 2;
  bf16_t* Ab  = (bf16_t*)(ws + off); off += (size_t)kB * kS * kD * 2;

  cvt_f32_bf16<<<4096, 256, 0, stream>>>(hs, Xb, kB * kS * kD);
  cvt_transpose_bf16<<<(3 * kD * kD + 255) / 256, 256, 0, stream>>>(Wc_attn, Wat, kD, 3 * kD);
  cvt_transpose_bf16<<<(kD * kD + 255) / 256, 256, 0, stream>>>(Wc_proj, Wpt, kD, kD);

  gemm_qkv<<<dim3(kB * kS / 16, 12), 128, 0, stream>>>(Xb, Wat, bc_attn, Qb, Kb, Vtb);
  flash_attn<<<(kB * kH * (kS / 16)) / 2, 64, 0, stream>>>(Qb, Kb, Vtb, amask, Ab);
  gemm_proj<<<dim3(kB * kS / 16, 4), 128, 0, stream>>>(Ab, Wpt, bc_proj, out);
}
